// GETSCalibrator_3109556322397
// MI455X (gfx1250) — matrix operand order repaired and checked by reading, not compiled, not run on hardware
//
#include <hip/hip_runtime.h>
#include <hip/hip_bf16.h>
#include <math.h>

// ---------------- constants from the reference ----------------
#define CNUM   64      // classes
#define FDIM   512     // feature dim
#define FH     32      // feature hidden
#define DH     16      // degree hidden
#define MAXDEG 128

typedef __attribute__((ext_vector_type(16))) __bf16 v16bf;
typedef __attribute__((ext_vector_type(8)))  float  v8f;

union AFrag { v16bf v; __bf16 e[16]; };
union CFrag { v8f   v; float  e[8];  };

// ---------------- 1) zero degree counters ----------------
__global__ void k_zero_counts(int* cnt_sd, int* cnt_d, int n) {
    int i = blockIdx.x * blockDim.x + threadIdx.x;
    if (i < n) { cnt_sd[i] = 0; cnt_d[i] = 0; }
}

// ---------------- 2) degree accumulation over edges ----------------
__global__ void k_degrees(const int* __restrict__ ei, int E,
                          int* cnt_sd, int* cnt_d) {
    int e = blockIdx.x * blockDim.x + threadIdx.x;
    if (e >= E) return;
    int s = ei[e];        // edge_index[0][e]
    int d = ei[E + e];    // edge_index[1][e]
    atomicAdd(&cnt_sd[s], 1);
    atomicAdd(&cnt_sd[d], 1);
    atomicAdd(&cnt_d[d], 1);
}

// ---------------- 3) dinv = rsqrt(deg_dst+1), deg_idx = clip ----------------
__global__ void k_dinv(const int* __restrict__ cnt_sd, const int* __restrict__ cnt_d,
                       float* dinv, int* degidx, int n) {
    int i = blockIdx.x * blockDim.x + threadIdx.x;
    if (i >= n) return;
    dinv[i] = rsqrtf((float)cnt_d[i] + 1.0f);
    int d = cnt_sd[i];
    degidx[i] = d < (MAXDEG - 1) ? d : (MAXDEG - 1);
}

// ---------------- 4) weights f32 row-major [512][32] -> bf16 K-major [32][512] ----------------
__global__ void k_cvt_weights(const float* __restrict__ Wf0, const float* __restrict__ Wf2,
                              const float* __restrict__ Wfg, __bf16* __restrict__ wB) {
    int idx = blockIdx.x * blockDim.x + threadIdx.x;   // 3 * 512 * 32
    if (idx >= 3 * FDIM * FH) return;
    int wi  = idx / (FDIM * FH);
    int r   = idx - wi * (FDIM * FH);
    int k   = r / FH;
    int col = r - k * FH;
    const float* W = (wi == 0) ? Wf0 : ((wi == 1) ? Wf2 : Wfg);
    wB[(size_t)wi * FDIM * FH + (size_t)col * FDIM + k] = (__bf16)W[k * FH + col];
}

// ---------------- 5) fused 3-projection WMMA GEMM: Pf[n,96] ----------------
// One wave = one 16-row tile of nodes; A fragment (features, bf16) is shared by
// all 3 weights and both 16-column output tiles: 6 wmma per K-step, 16 K-steps.
__global__ __launch_bounds__(256) void k_featproj(
        const float* __restrict__ feat, const __bf16* __restrict__ wB,
        const float* __restrict__ bf0, const float* __restrict__ bf2,
        const float* __restrict__ bfg, float* __restrict__ Pf, int N) {
    int wave = threadIdx.x >> 5;
    int lane = threadIdx.x & 31;
    int tile = blockIdx.x * 8 + wave;
    int ntiles = (N + 15) >> 4;
    if (tile >= ntiles) return;

    int m  = lane & 15;    // row within tile (A layout)
    int hh = lane >> 4;    // half-wave selector (K split / M offset)
    int row = tile * 16 + m;
    if (row >= N) row = N - 1;              // clamp (N is a multiple of 16 anyway)
    const float* frow = feat + (size_t)row * FDIM;

    CFrag acc[3][2];
    #pragma unroll
    for (int wi = 0; wi < 3; ++wi)
        #pragma unroll
        for (int t = 0; t < 2; ++t)
            #pragma unroll
            for (int r = 0; r < 8; ++r) acc[wi][t].e[r] = 0.0f;

    for (int kk = 0; kk < 16; ++kk) {
        int kbase = kk * 32;
        // ---- A fragment: 16x32 bf16, lane(m,hh) holds K = {16hh..16hh+15}
        const float4* p0 = (const float4*)(frow + kbase + 16 * hh);
        const float4* p1 = (const float4*)(frow + kbase + 8 + 16 * hh);
        float4 f0 = p0[0], f1 = p0[1];
        float4 f2 = p1[0], f3 = p1[1];
        AFrag a;
        a.e[0]=(__bf16)f0.x; a.e[1]=(__bf16)f0.y; a.e[2]=(__bf16)f0.z; a.e[3]=(__bf16)f0.w;
        a.e[4]=(__bf16)f1.x; a.e[5]=(__bf16)f1.y; a.e[6]=(__bf16)f1.z; a.e[7]=(__bf16)f1.w;
        a.e[8]=(__bf16)f2.x; a.e[9]=(__bf16)f2.y; a.e[10]=(__bf16)f2.z; a.e[11]=(__bf16)f2.w;
        a.e[12]=(__bf16)f3.x; a.e[13]=(__bf16)f3.y; a.e[14]=(__bf16)f3.z; a.e[15]=(__bf16)f3.w;

        if (kk + 1 < 16) __builtin_prefetch(frow + kbase + 32, 0, 0);  // global_prefetch_b8

        #pragma unroll
        for (int wi = 0; wi < 3; ++wi) {
            #pragma unroll
            for (int t = 0; t < 2; ++t) {
                // ---- B fragment: 32x16 bf16, lane holds column (lane&15)+16t, K-contiguous
                int col = (lane & 15) + 16 * t;
                const v16bf* bp = (const v16bf*)(wB + (size_t)wi * FDIM * FH
                                                    + (size_t)col * FDIM + kbase + 16 * hh);
                v16bf b = *bp;
                acc[wi][t].v = __builtin_amdgcn_wmma_f32_16x16x32_bf16(
                    false, a.v, false, b, (short)0, acc[wi][t].v, false, false);
            }
        }
    }

    // ---- write-out: C layout lane l, vgpr r -> M = r + 8*(l>>4), Ncol = l&15
    #pragma unroll
    for (int wi = 0; wi < 3; ++wi) {
        const float* bb = (wi == 0) ? bf0 : ((wi == 1) ? bf2 : bfg);
        #pragma unroll
        for (int t = 0; t < 2; ++t) {
            int col = t * 16 + (lane & 15);
            float bias = bb[col];
            #pragma unroll
            for (int r = 0; r < 8; ++r) {
                int orow = tile * 16 + r + 8 * hh;
                if (orow < N)
                    Pf[(size_t)orow * 96 + wi * 32 + col] = acc[wi][t].e[r] + bias;
            }
        }
    }
}

// ---------------- 6) per-node expert GEMMs + self term + gating ----------------
// 4 nodes per block, 64 threads/node (one per output class).
__global__ __launch_bounds__(256) void k_expert_gate(
        const float* __restrict__ logits, const float* __restrict__ Pf,
        const int* __restrict__ degidx, const float* __restrict__ dinv,
        const float* __restrict__ Emb1, const float* __restrict__ Emb2,
        const float* __restrict__ Embg,
        const float* __restrict__ W0, const float* __restrict__ b0,
        const float* __restrict__ W1, const float* __restrict__ b1,
        const float* __restrict__ W2, const float* __restrict__ b2,
        const float* __restrict__ wg,
        float* __restrict__ h, float* __restrict__ agg, float* __restrict__ gates, int N) {
    __shared__ float sx[4][208];   // [0,64) logits | [64,160) Pf(96) | [160,208) deg-embs(48)
    __shared__ float scl[4][3];
    int nl = threadIdx.x >> 6;
    int c  = threadIdx.x & 63;
    int node = blockIdx.x * 4 + nl;
    bool act = node < N;

    if (act) {
        int deg = degidx[node];
        for (int i = c; i < 208; i += 64) {
            float v;
            if (i < 64)       v = logits[(size_t)node * CNUM + i];
            else if (i < 160) v = Pf[(size_t)node * 96 + (i - 64)];
            else {
                int j = i - 160;
                v = (j < 16) ? Emb1[deg * DH + j]
                  : (j < 32) ? Emb2[deg * DH + (j - 16)]
                             : Embg[deg * DH + (j - 32)];
            }
            sx[nl][i] = v;
        }
    }
    __syncthreads();

    if (act) {
        const float* L  = sx[nl];
        const float* P0 = sx[nl] + 64;
        const float* P2 = sx[nl] + 96;
        const float* Pg = sx[nl] + 128;
        const float* E1 = sx[nl] + 160;
        const float* E2 = sx[nl] + 176;
        const float* Eg = sx[nl] + 192;

        float h0 = 0.f, h1 = 0.f, h2 = 0.f;
        #pragma unroll 8
        for (int k = 0; k < 64; ++k) {
            float lv = L[k];
            h0 += lv * W0[k * 64 + c];
            h1 += lv * W1[k * 64 + c];
        }
        #pragma unroll 8
        for (int k = 0; k < 32; ++k) {
            h0 += P0[k] * W0[(64 + k) * 64 + c];
            h2 += P2[k] * W2[k * 64 + c];
        }
        #pragma unroll
        for (int k = 0; k < 16; ++k) {
            h1 += E1[k] * W1[(64 + k) * 64 + c];
            h2 += E2[k] * W2[(32 + k) * 64 + c];
        }
        float dv = dinv[node];
        float d2 = dv * dv;
        size_t base = (size_t)node * 192;
        h[base + c]       = h0;
        h[base + 64 + c]  = h1;
        h[base + 128 + c] = h2;
        agg[base + c]       = h0 * d2 + b0[c];   // self-loop term + bias folded in
        agg[base + 64 + c]  = h1 * d2 + b1[c];
        agg[base + 128 + c] = h2 * d2 + b2[c];

        if (c < 3) {
            float s = 0.f;
            for (int k = 0; k < 64; ++k) s += L[k]  * wg[k * 3 + c];
            for (int k = 0; k < 32; ++k) s += Pg[k] * wg[(64 + k) * 3 + c];
            for (int k = 0; k < 16; ++k) s += Eg[k] * wg[(96 + k) * 3 + c];
            scl[nl][c] = s;
        }
    }
    __syncthreads();

    if (act && c == 0) {
        float c0 = scl[nl][0], c1 = scl[nl][1], c2 = scl[nl][2];
        // top-2 (first-index tie-break, like lax.top_k)
        int i1 = 0; float m1 = c0;
        if (c1 > m1) { m1 = c1; i1 = 1; }
        if (c2 > m1) { m1 = c2; i1 = 2; }
        int i2 = -1; float m2 = -3.0e38f;
        for (int j = 0; j < 3; ++j) {
            float vj = (j == 0) ? c0 : ((j == 1) ? c1 : c2);
            if (j != i1 && vj > m2) { m2 = vj; i2 = j; }
        }
        float eb = __expf(m2 - m1);    // softmax over [m1, m2]
        float inv = 1.0f / (1.0f + eb);
        float g[3] = {0.f, 0.f, 0.f};
        g[i1] = inv;
        g[i2] = eb * inv;
        gates[node * 4 + 0] = g[0];
        gates[node * 4 + 1] = g[1];
        gates[node * 4 + 2] = g[2];
        gates[node * 4 + 3] = 0.f;
    }
}

// ---------------- 7) edge scatter-aggregate (all 3 experts fused: 192 ch) ----------------
__global__ __launch_bounds__(256) void k_edge_agg(
        const int* __restrict__ ei, int E, const float* __restrict__ dinv,
        const float* __restrict__ h, float* __restrict__ agg) {
    int e = blockIdx.x * blockDim.x + threadIdx.x;
    if (e >= E) return;
    int s = ei[e];
    int d = ei[E + e];
    float coef = dinv[s] * dinv[d];
    const float4* hs = (const float4*)(h + (size_t)s * 192);
    float* ad = agg + (size_t)d * 192;
    #pragma unroll 4
    for (int i = 0; i < 48; ++i) {
        float4 v = hs[i];
        atomicAdd(ad + 4 * i + 0, v.x * coef);
        atomicAdd(ad + 4 * i + 1, v.y * coef);
        atomicAdd(ad + 4 * i + 2, v.z * coef);
        atomicAdd(ad + 4 * i + 3, v.w * coef);
    }
}

// ---------------- 8) gated combine + softplus temperature ----------------
__global__ __launch_bounds__(256) void k_final(
        const float* __restrict__ logits, const float* __restrict__ agg,
        const float* __restrict__ gates, float* __restrict__ out, int total) {
    int i = blockIdx.x * blockDim.x + threadIdx.x;
    if (i >= total) return;
    int n = i >> 6;
    int c = i & 63;
    size_t base = (size_t)n * 192;
    float g0 = gates[n * 4 + 0], g1 = gates[n * 4 + 1], g2 = gates[n * 4 + 2];
    float comb = g0 * agg[base + c] + g1 * agg[base + 64 + c] + g2 * agg[base + 128 + c];
    float sp = (comb > 20.f) ? comb : log1pf(__expf(comb));   // softplus
    out[i] = logits[i] * sp;
}

// ---------------- launcher ----------------
static inline size_t align256(size_t x) { return (x + 255) & ~(size_t)255; }

extern "C" void kernel_launch(void* const* d_in, const int* in_sizes, int n_in,
                              void* d_out, int out_size, void* d_ws, size_t ws_size,
                              hipStream_t stream) {
    const float* logits = (const float*)d_in[0];
    const float* feat   = (const float*)d_in[1];
    const int*   ei     = (const int*)  d_in[2];
    const float* Wf0    = (const float*)d_in[3];
    const float* bf0    = (const float*)d_in[4];
    const float* W0     = (const float*)d_in[5];
    const float* b0     = (const float*)d_in[6];
    const float* Emb1   = (const float*)d_in[7];
    const float* W1     = (const float*)d_in[8];
    const float* b1     = (const float*)d_in[9];
    const float* Wf2    = (const float*)d_in[10];
    const float* bf2    = (const float*)d_in[11];
    const float* Emb2   = (const float*)d_in[12];
    const float* W2     = (const float*)d_in[13];
    const float* b2     = (const float*)d_in[14];
    const float* Wfg    = (const float*)d_in[15];
    const float* bfg    = (const float*)d_in[16];
    const float* Embg   = (const float*)d_in[17];
    const float* wg     = (const float*)d_in[18];
    float* out = (float*)d_out;

    const int N = in_sizes[0] / CNUM;
    const int E = in_sizes[2] / 2;

    // workspace carve-out
    char* ws = (char*)d_ws;
    size_t o = 0;
    int*   cnt_sd = (int*)(ws + o);   o += align256((size_t)N * 4);
    int*   cnt_d  = (int*)(ws + o);   o += align256((size_t)N * 4);
    int*   degidx = (int*)(ws + o);   o += align256((size_t)N * 4);
    float* dinv   = (float*)(ws + o); o += align256((size_t)N * 4);
    float* Pf     = (float*)(ws + o); o += align256((size_t)N * 96 * 4);
    __bf16* wB    = (__bf16*)(ws + o); o += align256((size_t)3 * FDIM * FH * 2);
    float* h      = (float*)(ws + o); o += align256((size_t)N * 192 * 4);
    float* agg    = (float*)(ws + o); o += align256((size_t)N * 192 * 4);
    float* gates  = (float*)(ws + o); o += align256((size_t)N * 4 * 4);
    (void)ws_size; (void)n_in; (void)out_size;

    const int B = 256;
    k_zero_counts<<<(N + B - 1) / B, B, 0, stream>>>(cnt_sd, cnt_d, N);
    k_degrees<<<(E + B - 1) / B, B, 0, stream>>>(ei, E, cnt_sd, cnt_d);
    k_dinv<<<(N + B - 1) / B, B, 0, stream>>>(cnt_sd, cnt_d, dinv, degidx, N);
    k_cvt_weights<<<(3 * FDIM * FH + B - 1) / B, B, 0, stream>>>(Wf0, Wf2, Wfg, wB);

    int ntiles = (N + 15) / 16;
    k_featproj<<<(ntiles + 7) / 8, B, 0, stream>>>(feat, wB, bf0, bf2, bfg, Pf, N);

    k_expert_gate<<<(N + 3) / 4, B, 0, stream>>>(
        logits, Pf, degidx, dinv, Emb1, Emb2, Embg,
        W0, b0, W1, b1, W2, b2, wg, h, agg, gates, N);

    k_edge_agg<<<(E + B - 1) / B, B, 0, stream>>>(ei, E, dinv, h, agg);

    k_final<<<((size_t)N * CNUM + B - 1) / B, B, 0, stream>>>(logits, agg, gates, out, N * CNUM);
}
